// SimplifiedPrecisionAttentionBlock_87290915324058
// MI455X (gfx1250) — matrix operand order repaired and checked by reading, not compiled, not run on hardware
//
#include <hip/hip_runtime.h>
#include <hip/hip_bf16.h>
#include <math.h>

#define NSEQ 512
#define DDIM 64
#define BB 2
#define EPSF 1e-8f

typedef __attribute__((ext_vector_type(16))) _Float16 v16h;
typedef __attribute__((ext_vector_type(8)))  float    v8f;

// ---------------- output offsets (floats, return-order concat) ----------------
#define EST_OFF 0L
#define OUT_OFF ((long)BB*2*NSEQ*DDIM)                    // 131072
#define Q_OFF   (OUT_OFF + (long)BB*2*NSEQ*DDIM)         // 262144
#define Z_OFF   (Q_OFF + (long)BB*2*NSEQ*NSEQ)           // 1310720
#define LAM_OFF (Z_OFF + (long)BB*2*NSEQ*NSEQ*DDIM)      // 68419584

// ---------------- workspace offsets (floats) ----------------
#define WS_EFR 0L
#define WS_EFI (WS_EFR + (long)NSEQ*DDIM)
#define WS_W2  (WS_EFI + (long)NSEQ*DDIM)
#define WS_T   (WS_W2  + (long)NSEQ*DDIM)
#define WS_UQR (WS_T   + NSEQ)
#define WS_UQI (WS_UQR + (long)BB*NSEQ*DDIM)
#define WS_UKR (WS_UQI + (long)BB*NSEQ*DDIM)
#define WS_UKI (WS_UKR + (long)BB*NSEQ*DDIM)
#define WS_UVR (WS_UKI + (long)BB*NSEQ*DDIM)
#define WS_UVI (WS_UVR + (long)BB*NSEQ*DDIM)
#define WS_VR  (WS_UVI + (long)BB*NSEQ*DDIM)
#define WS_VI  (WS_VR  + (long)BB*NSEQ*DDIM)
#define WS_A   (WS_VI  + (long)BB*NSEQ*DDIM)
#define WS_PR  (WS_A   + (long)BB*NSEQ*NSEQ)
#define WS_PI  (WS_PR  + (long)BB*NSEQ*DDIM)

// ================= WMMA helpers (wave32, 16x16x32 f16 -> f32) =================
static __device__ __forceinline__ v8f wmma_f16(v16h a, v16h b, v8f c) {
  return __builtin_amdgcn_wmma_f32_16x16x32_f16(false, a, false, b, (short)0, c,
                                                false, false);
}

// A tile 16x32 (M x K), source row-major f32, row stride ld.
// Layout: lanes 0-15 -> M=lane, K 0..15; lanes 16-31 -> K 16..31 (contiguous, matches B)
static __device__ __forceinline__ v16h ld_a(const float* s, int ld, int lane) {
  int half = lane >> 4, m = lane & 15;
  v16h a;
#pragma unroll
  for (int e = 0; e < 16; ++e) {
    int vg = e >> 1, lo = e & 1;
    int k = vg * 2 + lo + half * 16;
    a[e] = (_Float16)s[m * ld + k];
  }
  return a;
}
static __device__ __forceinline__ v16h ld_a_neg(const float* s, int ld, int lane) {
  int half = lane >> 4, m = lane & 15;
  v16h a;
#pragma unroll
  for (int e = 0; e < 16; ++e) {
    int vg = e >> 1, lo = e & 1;
    int k = vg * 2 + lo + half * 16;
    a[e] = (_Float16)(-s[m * ld + k]);
  }
  return a;
}
static __device__ __forceinline__ v16h ld_a_mul(const float* s, const float* w,
                                                int ld, int lane) {
  int half = lane >> 4, m = lane & 15;
  v16h a;
#pragma unroll
  for (int e = 0; e < 16; ++e) {
    int vg = e >> 1, lo = e & 1;
    int k = vg * 2 + lo + half * 16;
    a[e] = (_Float16)(s[m * ld + k] * w[m * ld + k]);
  }
  return a;
}
// B tile 32x16 (K x N) from a TRANSPOSED row-major source: B[k][n] = s[n*ld + k]
static __device__ __forceinline__ v16h ld_bT(const float* s, int ld, int lane) {
  int half = lane >> 4, n = lane & 15;
  v16h b;
#pragma unroll
  for (int e = 0; e < 16; ++e) {
    int k = e + half * 16;
    b[e] = (_Float16)s[n * ld + k];
  }
  return b;
}
// B tile with on-the-fly |.|^2: B[k][n] = sr[n*ld+k]^2 + si[n*ld+k]^2
static __device__ __forceinline__ v16h ld_bT_sq(const float* sr, const float* si,
                                                int ld, int lane) {
  int half = lane >> 4, n = lane & 15;
  v16h b;
#pragma unroll
  for (int e = 0; e < 16; ++e) {
    int k = e + half * 16;
    float r = sr[n * ld + k], i = si[n * ld + k];
    b[e] = (_Float16)(r * r + i * i);
  }
  return b;
}
// B tile from row-major source: B[k][n] = s[k*ld + n]
static __device__ __forceinline__ v16h ld_bR(const float* s, int ld, int lane) {
  int half = lane >> 4, n = lane & 15;
  v16h b;
#pragma unroll
  for (int e = 0; e < 16; ++e) {
    int k = e + half * 16;
    b[e] = (_Float16)s[k * ld + n];
  }
  return b;
}

// ================= K1: phases, W2, t, lambda_h =================
__global__ __launch_bounds__(256) void setup_kernel(const float* __restrict__ tm,
                                                    const float* __restrict__ lam_half,
                                                    float* __restrict__ ws,
                                                    float* __restrict__ lam_out) {
  int idx = blockIdx.x * 256 + threadIdx.x;
  float denom = tm[NSEQ - 1] - tm[0];
  if (idx < NSEQ * DDIM) {
    int i = idx / DDIM, d = idx % DDIM;
    float t = tm[i] / denom;
    float lam = (d < DDIM / 2) ? lam_half[d] : -lam_half[d - DDIM / 2];
    float ph = t * lam;
    float cr = cosf(ph), si = sinf(ph);
    ws[WS_EFR + idx] = cr;
    ws[WS_EFI + idx] = si;
    ws[WS_W2 + idx] = cr * cr + si * si;
  }
  if (idx < NSEQ) ws[WS_T + idx] = tm[idx] / denom;
  if (idx < DDIM) {
    float lam = (idx < DDIM / 2) ? lam_half[idx] : -lam_half[idx - DDIM / 2];
    lam_out[idx] = 0.f;
    lam_out[DDIM + idx] = lam;
  }
}

// ================= K2/K5: complex linear projection (WMMA) =================
// yr = Zr@W0^T - Zi@W1^T + b0 ; yi = Zr@W1^T + Zi@W0^T + b1
// do_rot: multiply by conj(ef) and store U (and raw V if sR!=null); else store plain.
__global__ __launch_bounds__(32) void proj_kernel(
    const float* __restrict__ Zr_base, const float* __restrict__ Zi_base, long zstride,
    const float* __restrict__ W, const float* __restrict__ bias,
    const float* __restrict__ efr, const float* __restrict__ efi,
    float* __restrict__ pR_base, float* __restrict__ pI_base, long pstride,
    float* __restrict__ sR, float* __restrict__ sI, int do_rot) {
  int lane = threadIdx.x & 31;
  int b = blockIdx.x >> 5;
  int it = blockIdx.x & 31;
  int row0 = it * 16;
  const float* Zr = Zr_base + (long)b * zstride;
  const float* Zi = Zi_base + (long)b * zstride;
  const float* W0 = W;
  const float* W1 = W + DDIM * DDIM;
  int half = lane >> 4, n = lane & 15;
#pragma unroll
  for (int dt = 0; dt < 4; ++dt) {
    int d0 = dt * 16;
    v8f yr = {}; v8f yi = {};
#pragma unroll
    for (int kc = 0; kc < DDIM; kc += 32) {
      v16h aR = ld_a(Zr + row0 * DDIM + kc, DDIM, lane);
      v16h aI = ld_a(Zi + row0 * DDIM + kc, DDIM, lane);
      v16h aIn = ld_a_neg(Zi + row0 * DDIM + kc, DDIM, lane);
      v16h b0 = ld_bT(W0 + d0 * DDIM + kc, DDIM, lane);
      v16h b1 = ld_bT(W1 + d0 * DDIM + kc, DDIM, lane);
      yr = wmma_f16(aR, b0, yr);
      yr = wmma_f16(aIn, b1, yr);
      yi = wmma_f16(aR, b1, yi);
      yi = wmma_f16(aI, b0, yi);
    }
    int d = d0 + n;
    float b0v = bias[d], b1v = bias[DDIM + d];
#pragma unroll
    for (int r = 0; r < 8; ++r) {
      int nr = row0 + r + half * 8;
      float qr = yr[r] + b0v;
      float qi = yi[r] + b1v;
      long pidx = (long)b * pstride + (long)nr * DDIM + d;
      if (do_rot) {
        float er = efr[nr * DDIM + d], ei = efi[nr * DDIM + d];
        pR_base[pidx] = er * qr + ei * qi;   // conj(ef) * q, real
        pI_base[pidx] = er * qi - ei * qr;   // conj(ef) * q, imag
        if (sR) {
          long vidx = (long)b * NSEQ * DDIM + (long)nr * DDIM + d;
          sR[vidx] = qr;
          sI[vidx] = qi;
        }
      } else {
        pR_base[pidx] = qr;
        pI_base[pidx] = qi;
      }
    }
  }
}

// ================= K3: scores (WMMA) + causal softmax =================
__global__ __launch_bounds__(32) void scores_kernel(
    const float* __restrict__ wsc, float* __restrict__ wsm, float* __restrict__ out,
    const float* __restrict__ lw, const float* __restrict__ lg,
    const float* __restrict__ nf, const float* __restrict__ tauv) {
  __shared__ float sc[16 * NSEQ];
  __shared__ float t1s[16];
  int lane = threadIdx.x & 31;
  int it = blockIdx.x, b = blockIdx.y;
  const float* W2 = wsc + WS_W2;
  const float* tt = wsc + WS_T;
  const float* Uqr = wsc + WS_UQR + (long)b * NSEQ * DDIM;
  const float* Uqi = wsc + WS_UQI + (long)b * NSEQ * DDIM;
  const float* Ukr = wsc + WS_UKR + (long)b * NSEQ * DDIM;
  const float* Uki = wsc + WS_UKI + (long)b * NSEQ * DDIM;
  float lamW = lw[0] * lw[0];
  float lamG = lg[0] * lg[0] + EPSF;
  float nf2 = nf[0] * nf[0] + EPSF;
  float tau2 = tauv[0] * tauv[0];
  // t1[i] = sum_d W2[i,d]*(Uqr^2+Uqi^2)
  {
    int m = lane & 15, dh = (lane >> 4) * 32;
    const float* w2r = W2 + (long)(it * 16 + m) * DDIM + dh;
    const float* qr = Uqr + (long)(it * 16 + m) * DDIM + dh;
    const float* qi = Uqi + (long)(it * 16 + m) * DDIM + dh;
    float p = 0.f;
#pragma unroll
    for (int d = 0; d < 32; ++d) {
      float ar = qr[d], ai = qi[d];
      p += w2r[d] * (ar * ar + ai * ai);
    }
    p += __shfl_xor(p, 16, 32);
    if (lane < 16) t1s[lane] = p;
  }
  __syncthreads();
  int half = lane >> 4, nn = lane & 15;
  for (int jt = 0; jt <= it; ++jt) {
    v8f t3 = {}; v8f t2 = {};
#pragma unroll
    for (int kc = 0; kc < DDIM; kc += 32) {
      const float* w2p = W2 + (long)it * 16 * DDIM + kc;
      v16h aQr = ld_a_mul(Uqr + (long)it * 16 * DDIM + kc, w2p, DDIM, lane);
      v16h aQi = ld_a_mul(Uqi + (long)it * 16 * DDIM + kc, w2p, DDIM, lane);
      v16h aW2 = ld_a(w2p, DDIM, lane);
      v16h bKr = ld_bT(Ukr + (long)jt * 16 * DDIM + kc, DDIM, lane);
      v16h bKi = ld_bT(Uki + (long)jt * 16 * DDIM + kc, DDIM, lane);
      v16h bSq = ld_bT_sq(Ukr + (long)jt * 16 * DDIM + kc,
                          Uki + (long)jt * 16 * DDIM + kc, DDIM, lane);
      t3 = wmma_f16(aQr, bKr, t3);
      t3 = wmma_f16(aQi, bKi, t3);
      t2 = wmma_f16(aW2, bSq, t2);
    }
    int j = jt * 16 + nn;
    float tj = tt[j];
#pragma unroll
    for (int r = 0; r < 8; ++r) {
      int rI = r + half * 8;
      int i = it * 16 + rI;
      float R = t1s[rI] + t2[r] - 2.f * t3[r];
      float va = lamW * fabsf(tt[i] - tj) + lamG;
      float s = -tau2 * logf(nf2 * va + R);
      if (j > i) s = -INFINITY;
      sc[rI * NSEQ + j] = s;
    }
  }
  __syncthreads();
  int jmax = (it + 1) * 16;
  float* Abase = wsm + WS_A + (long)b * NSEQ * NSEQ;
  for (int r = 0; r < 16; ++r) {
    int i = it * 16 + r;
    float m = -INFINITY;
    for (int j = lane; j < jmax; j += 32) m = fmaxf(m, sc[r * NSEQ + j]);
#pragma unroll
    for (int off = 16; off; off >>= 1) m = fmaxf(m, __shfl_xor(m, off, 32));
    float ssum = 0.f;
    for (int j = lane; j < jmax; j += 32) ssum += __expf(sc[r * NSEQ + j] - m);
#pragma unroll
    for (int off = 16; off; off >>= 1) ssum += __shfl_xor(ssum, off, 32);
    float inv = 1.f / ssum;
    float* Arow = Abase + (long)i * NSEQ;
    float* qR = out + Q_OFF + ((long)(b * 2 + 0) * NSEQ + i) * NSEQ;
    float* qI = out + Q_OFF + ((long)(b * 2 + 1) * NSEQ + i) * NSEQ;
    for (int j = lane; j < NSEQ; j += 32) {
      float a = (j < jmax) ? __expf(sc[r * NSEQ + j] - m) * inv : 0.f;
      Arow[j] = a;
      qR[j] = a;
      qI[j] = 0.f;
    }
  }
}

// ================= K4: A@Uv (WMMA), rotation, gating, P =================
__global__ __launch_bounds__(32) void apply_kernel(const float* __restrict__ wsc,
                                                   float* __restrict__ wsm,
                                                   float* __restrict__ est_out,
                                                   const float* __restrict__ eta_p) {
  int lane = threadIdx.x & 31;
  int it = blockIdx.x, b = blockIdx.y;
  const float* A = wsc + WS_A + (long)b * NSEQ * NSEQ;
  const float* Uvr = wsc + WS_UVR + (long)b * NSEQ * DDIM;
  const float* Uvi = wsc + WS_UVI + (long)b * NSEQ * DDIM;
  const float* Vr = wsc + WS_VR + (long)b * NSEQ * DDIM;
  const float* Vi = wsc + WS_VI + (long)b * NSEQ * DDIM;
  const float* efr = wsc + WS_EFR;
  const float* efi = wsc + WS_EFI;
  int half = lane >> 4, nn = lane & 15;
  int jmax = (it + 1) * 16;  // causal: A rows are zero beyond this
#pragma unroll
  for (int dt = 0; dt < 4; ++dt) {
    int d0 = dt * 16;
    v8f er = {}; v8f ei = {};
    for (int jc = 0; jc < jmax; jc += 32) {
      v16h aA = ld_a(A + (long)it * 16 * NSEQ + jc, NSEQ, lane);
      v16h bR = ld_bR(Uvr + (long)jc * DDIM + d0, DDIM, lane);
      v16h bI = ld_bR(Uvi + (long)jc * DDIM + d0, DDIM, lane);
      er = wmma_f16(aA, bR, er);
      ei = wmma_f16(aA, bI, ei);
    }
    int d = d0 + nn;
    float eta = 1.f / (1.f + __expf(-eta_p[d]));
    float g = 1.f / (1.f + __expf(-eta));
#pragma unroll
    for (int r = 0; r < 8; ++r) {
      int nr = it * 16 + r + half * 8;
      float fr = efr[nr * DDIM + d], fi = efi[nr * DDIM + d];
      float estr = fr * er[r] - fi * ei[r];
      float esti = fr * ei[r] + fi * er[r];
      float vr = Vr[(long)nr * DDIM + d], vi = Vi[(long)nr * DDIM + d];
      float elr = (1.f - eta) * vr + g * estr;
      float eli = (1.f - eta) * vi + g * esti;
      est_out[((long)(b * 2 + 0) * NSEQ + nr) * DDIM + d] = elr;
      est_out[((long)(b * 2 + 1) * NSEQ + nr) * DDIM + d] = eli;
      long pidx = (long)b * NSEQ * DDIM + (long)nr * DDIM + d;
      wsm[WS_PR + pidx] = fr * elr - fi * eli;
      wsm[WS_PI + pidx] = fr * eli + fi * elr;
    }
  }
}

// ================= K6: Z_ij_hat_all broadcast product (bandwidth-bound) ======
__global__ __launch_bounds__(256) void zij_kernel(const float* __restrict__ ws,
                                                  float* __restrict__ out) {
  long idx = (long)blockIdx.x * 256 + threadIdx.x;
  const long JD = (long)NSEQ * (DDIM / 4);
  long jd = idx % JD;
  long t = idx / JD;
  int i = (int)(t % NSEQ);
  int b = (int)(t / NSEQ);
  int j = (int)(jd / (DDIM / 4));
  int d = (int)(jd % (DDIM / 4)) * 4;
  const float4 er = *(const float4*)(ws + WS_EFR + (long)i * DDIM + d);
  const float4 ei = *(const float4*)(ws + WS_EFI + (long)i * DDIM + d);
  const float4 vr = *(const float4*)(ws + WS_UVR + ((long)b * NSEQ + j) * DDIM + d);
  const float4 vi = *(const float4*)(ws + WS_UVI + ((long)b * NSEQ + j) * DDIM + d);
  float4 zr, zi;
  zr.x = er.x * vr.x - ei.x * vi.x;  zi.x = er.x * vi.x + ei.x * vr.x;
  zr.y = er.y * vr.y - ei.y * vi.y;  zi.y = er.y * vi.y + ei.y * vr.y;
  zr.z = er.z * vr.z - ei.z * vi.z;  zi.z = er.z * vi.z + ei.z * vr.z;
  zr.w = er.w * vr.w - ei.w * vi.w;  zi.w = er.w * vi.w + ei.w * vr.w;
  long or_ = Z_OFF + (((long)(b * 2 + 0) * NSEQ + i) * NSEQ + j) * DDIM + d;
  long oi_ = Z_OFF + (((long)(b * 2 + 1) * NSEQ + i) * NSEQ + j) * DDIM + d;
  *(float4*)(out + or_) = zr;
  *(float4*)(out + oi_) = zi;
}

// ================= host launcher =================
extern "C" void kernel_launch(void* const* d_in, const int* in_sizes, int n_in,
                              void* d_out, int out_size, void* d_ws, size_t ws_size,
                              hipStream_t stream) {
  const float* Zq  = (const float*)d_in[0];
  const float* Zk  = (const float*)d_in[1];
  const float* Zv  = (const float*)d_in[2];
  const float* tm  = (const float*)d_in[3];
  const float* Wqw = (const float*)d_in[4];
  const float* Wqb = (const float*)d_in[5];
  const float* Wkw = (const float*)d_in[6];
  const float* Wkb = (const float*)d_in[7];
  const float* Wvw = (const float*)d_in[8];
  const float* Wvb = (const float*)d_in[9];
  const float* Wpw = (const float*)d_in[10];
  const float* Wpb = (const float*)d_in[11];
  const float* lamh = (const float*)d_in[12];
  const float* lw  = (const float*)d_in[13];
  const float* lg  = (const float*)d_in[14];
  const float* nf  = (const float*)d_in[15];
  const float* tau = (const float*)d_in[16];
  const float* eta = (const float*)d_in[17];
  float* out = (float*)d_out;
  float* ws = (float*)d_ws;
  const long ND = (long)NSEQ * DDIM;

  setup_kernel<<<(NSEQ * DDIM + 255) / 256, 256, 0, stream>>>(tm, lamh, ws,
                                                              out + LAM_OFF);
  // Q, K, V projections (rotated by conj(ef)); V also stores raw values.
  proj_kernel<<<BB * 32, 32, 0, stream>>>(Zq, Zq + ND, 2 * ND, Wqw, Wqb,
                                          ws + WS_EFR, ws + WS_EFI,
                                          ws + WS_UQR, ws + WS_UQI, ND,
                                          nullptr, nullptr, 1);
  proj_kernel<<<BB * 32, 32, 0, stream>>>(Zk, Zk + ND, 2 * ND, Wkw, Wkb,
                                          ws + WS_EFR, ws + WS_EFI,
                                          ws + WS_UKR, ws + WS_UKI, ND,
                                          nullptr, nullptr, 1);
  proj_kernel<<<BB * 32, 32, 0, stream>>>(Zv, Zv + ND, 2 * ND, Wvw, Wvb,
                                          ws + WS_EFR, ws + WS_EFI,
                                          ws + WS_UVR, ws + WS_UVI, ND,
                                          ws + WS_VR, ws + WS_VI, 1);
  scores_kernel<<<dim3(NSEQ / 16, BB), 32, 0, stream>>>(ws, ws, out, lw, lg, nf, tau);
  apply_kernel<<<dim3(NSEQ / 16, BB), 32, 0, stream>>>(ws, ws, out + EST_OFF, eta);
  // Output projection (plain, interleaved complex layout in d_out).
  proj_kernel<<<BB * 32, 32, 0, stream>>>(ws + WS_PR, ws + WS_PI, ND, Wpw, Wpb,
                                          ws + WS_EFR, ws + WS_EFI,
                                          out + OUT_OFF, out + OUT_OFF + ND, 2 * ND,
                                          nullptr, nullptr, 0);
  // 268 MB broadcast-product output: pure store-bandwidth kernel.
  long zthreads = (long)BB * NSEQ * NSEQ * (DDIM / 4);
  zij_kernel<<<(unsigned)(zthreads / 256), 256, 0, stream>>>(ws, out);
}
